// MultiHeadAttention_25769804194
// MI455X (gfx1250) — matrix operand order repaired and checked by reading, not compiled, not run on hardware
//
#include <hip/hip_runtime.h>
#include <hip/hip_bf16.h>
#include <math.h>
#include <stdint.h>

typedef __attribute__((ext_vector_type(16))) __bf16 v16bf;
typedef __attribute__((ext_vector_type(8)))  __bf16 v8bf;
typedef __attribute__((ext_vector_type(8)))  float  v8f;
typedef __attribute__((ext_vector_type(4)))  unsigned int v4u;
typedef __attribute__((ext_vector_type(8)))  int v8i;
typedef __attribute__((ext_vector_type(4)))  int v4i;

#define HN     16
#define DK     64
#define DMODEL 1024
#define LSEQ   2048
#define BB     2

static __device__ __forceinline__ v16bf cat8(v8bf lo, v8bf hi) {
  v16bf r;
#pragma unroll
  for (int i = 0; i < 8; ++i) { r[i] = lo[i]; r[i + 8] = hi[i]; }
  return r;
}

// Low 32 bits of a generic pointer to LDS == byte offset within the
// workgroup's LDS allocation (ISA 10.2 aperture mapping).
static __device__ __forceinline__ unsigned lds_offset(const void* p) {
  return (unsigned)(unsigned long long)(uintptr_t)p;
}

// ---------------------------------------------------------------------------
// Tensor Data Mover: async 2D tile (tile_w x tile_h elements, bf16) from
// global to LDS.  D# bitfields per CDNA5 ISA ch.8 (group0/group1; groups 2-3
// zero => 2D tensor).  Tracked by TENSORcnt.  6-arg builtin form
// (g0, g1, g2, g3, g4, cpol).
// ---------------------------------------------------------------------------
static __device__ __forceinline__ void tdm_load_2d_bf16(
    unsigned lds_addr, const void* gptr, unsigned tile_w, unsigned tile_h,
    unsigned long long row_stride_elems) {
  const unsigned long long ga = (unsigned long long)(uintptr_t)gptr;
  v4u g0;
  g0[0] = 1u;                                   // count=1, user descriptor
  g0[1] = lds_addr;                             // LDS byte address
  g0[2] = (unsigned)ga;                         // global_addr[31:0]
  g0[3] = (unsigned)(ga >> 32) | (2u << 30);    // global_addr[56:32] | type=2
  v8i g1;
  g1[0] = (int)(1u << 16);                      // workgroup_mask=0, data_size=1 (2B)
  g1[1] = (int)((tile_w & 0xFFFFu) << 16);      // tensor_dim0[15:0]
  g1[2] = (int)((tile_h & 0xFFFFu) << 16);      // tensor_dim0[31:16]=0 | tensor_dim1[15:0]
  g1[3] = (int)((tile_w & 0xFFFFu) << 16);      // tensor_dim1[31:16]=0 | tile_dim0
  g1[4] = (int)(tile_h & 0xFFFFu);              // tile_dim1 | tile_dim2=0
  g1[5] = (int)(unsigned)row_stride_elems;      // tensor_dim0_stride[31:0]
  g1[6] = (int)(unsigned)(row_stride_elems >> 32); // stride[47:32] | stride1[15:0]=0
  g1[7] = 0;
  const v4i z4 = {0, 0, 0, 0};
  const v8i z8 = {0, 0, 0, 0, 0, 0, 0, 0};
  __builtin_amdgcn_tensor_load_to_lds(g0, g1, z4, z4, z8, 0);
}

// ---------------------------------------------------------------------------
// Elementwise f32 -> bf16 conversion (one pass; all GEMMs then run bf16 WMMA)
// ---------------------------------------------------------------------------
__global__ void f32_to_bf16_kernel(const float* __restrict__ src,
                                   __bf16* __restrict__ dst, int n) {
  int i = blockIdx.x * blockDim.x + threadIdx.x;
  int stride = gridDim.x * blockDim.x;
  for (; i < n; i += stride) dst[i] = (__bf16)src[i];
}

// ---------------------------------------------------------------------------
// Projection GEMM: out[m,n] = sum_k X[m,k]*W[n,k] + bias[n]
// One wave computes a 16(M) x 64(N) strip = one head's columns.
// MODE 0/1: +RoPE, store bf16 (B,H,L,Dk)   (query / key)
// MODE 2: store bf16 transposed (B,H,Dk,L) (value)
// MODE 3: store f32 (B*L, D) to d_out      (output projection)
// ---------------------------------------------------------------------------
template <int MODE>
__global__ void proj_kernel(const __bf16* __restrict__ X,
                            const __bf16* __restrict__ W,
                            const float* __restrict__ bias,
                            void* __restrict__ out) {
  const int wid  = blockIdx.x * (blockDim.x >> 5) + (threadIdx.x >> 5);
  const int lane = threadIdx.x & 31;
  const int li   = lane & 15;
  const int half = lane >> 4;
  const int m0   = (wid >> 4) << 4;   // 256 m-tiles over B*L rows
  const int h    = wid & 15;          // 16 column strips (heads)
  const int n0   = h * DK;

  v8f acc[4];
#pragma unroll
  for (int t = 0; t < 4; ++t)
#pragma unroll
    for (int r = 0; r < 8; ++r) acc[t][r] = 0.f;

  const __bf16* xrow = X + (size_t)(m0 + li) * DMODEL;

  for (int kk = 0; kk < DMODEL; kk += 32) {
    // A tile: row M = li, K chunks at half*8 and 16+half*8
    v16bf a = cat8(*(const v8bf*)(xrow + kk + half * 8),
                   *(const v8bf*)(xrow + kk + half * 8 + 16));
#pragma unroll
    for (int t = 0; t < 4; ++t) {
      // B tile: col N = li, K chunks at half*8 and 16+half*8
      const __bf16* wrow = W + (size_t)(n0 + t * 16 + li) * DMODEL + kk + half * 8;
      v16bf b = cat8(*(const v8bf*)wrow, *(const v8bf*)(wrow + 16));
      acc[t] = __builtin_amdgcn_wmma_f32_16x16x32_bf16(
          false, a, false, b, (short)0, acc[t], false, false);
    }
  }

#pragma unroll
  for (int t = 0; t < 4; ++t) {
    const float bi = bias[n0 + t * 16 + li];
#pragma unroll
    for (int r = 0; r < 8; ++r) acc[t][r] += bi;
  }

  const int bidx = m0 / LSEQ;
  const int pos0 = m0 % LSEQ;

  if (MODE == 0 || MODE == 1) {
    // RoPE: pairs (dk, dk+32) live in acc[t] / acc[t+2], same lane & register.
#pragma unroll
    for (int t = 0; t < 2; ++t) {
      const int j = t * 16 + li;  // 0..31
      const float invf = __expf(-(float)j * 0.28782313662425572f);  // ln(1e4)/32
#pragma unroll
      for (int r = 0; r < 8; ++r) {
        float s, c;
        __sincosf((float)(pos0 + r + 8 * half) * invf, &s, &c);
        const float av = acc[t][r], bv = acc[t + 2][r];
        acc[t][r]     = av * c - bv * s;
        acc[t + 2][r] = bv * c + av * s;
      }
    }
  }

  if (MODE == 3) {
    float* o = (float*)out;
#pragma unroll
    for (int t = 0; t < 4; ++t)
#pragma unroll
      for (int r = 0; r < 8; ++r)
        o[(size_t)(m0 + r + 8 * half) * DMODEL + n0 + t * 16 + li] = acc[t][r];
  } else if (MODE == 2) {
    __bf16* o = (__bf16*)out;  // (B,H,Dk,L)
#pragma unroll
    for (int t = 0; t < 4; ++t) {
      const int dk = t * 16 + li;
#pragma unroll
      for (int r = 0; r < 8; ++r)
        o[((size_t)(bidx * HN + h) * DK + dk) * LSEQ + pos0 + r + 8 * half] =
            (__bf16)acc[t][r];
    }
  } else {
    __bf16* o = (__bf16*)out;  // (B,H,L,Dk)
#pragma unroll
    for (int t = 0; t < 4; ++t) {
      const int dk = t * 16 + li;
#pragma unroll
      for (int r = 0; r < 8; ++r)
        o[((size_t)(bidx * HN + h) * LSEQ + pos0 + r + 8 * half) * DK + dk] =
            (__bf16)acc[t][r];
    }
  }
}

// ---------------------------------------------------------------------------
// Fused attention: one wave per (b, h, 16-row q block); the 4 waves of a
// block share (b,h), so K/V tiles are staged once per block into LDS by the
// Tensor Data Mover (double-buffered, TENSORcnt-synchronized).
// Phase 1: QK^T tiles -> scale/mask/bias -> raw logits written to their final
//          d_out slot (L2-hot spill), online (max,sum) softmax per row.
// Phase 2: read slab back, overwrite with normalized weights (final output),
//          and accumulate P @ V^T with WMMA.
// ---------------------------------------------------------------------------
__global__ void attn_kernel(const __bf16* __restrict__ qg,
                            const __bf16* __restrict__ kg,
                            const __bf16* __restrict__ vt,
                            const int* __restrict__ mask,
                            const float* __restrict__ bias,
                            float* __restrict__ wout,
                            __bf16* __restrict__ attn) {
  __shared__ __align__(16) __bf16 kbuf[2][16 * DK];   // 2 x 2KB K tiles
  __shared__ __align__(16) __bf16 vbuf[2][DK * 32];   // 2 x 4KB V tiles
  __shared__ float smax[4][16];
  __shared__ float sinv[4][16];

  const int wslot = threadIdx.x >> 5;
  const int wid   = blockIdx.x * (blockDim.x >> 5) + wslot;
  const int lane  = threadIdx.x & 31;
  const int li    = lane & 15;
  const int half  = lane >> 4;
  const int b     = wid >> 11;
  const int h     = (wid >> 7) & 15;
  const int q0    = (wid & 127) << 4;

  // A operand: q rows (full Dk=64 per row, loaded once)
  const __bf16* qrow = qg + ((size_t)(b * HN + h) * LSEQ + q0 + li) * DK;
  const v16bf a0 = cat8(*(const v8bf*)(qrow + half * 8),
                        *(const v8bf*)(qrow + 16 + half * 8));
  const v16bf a1 = cat8(*(const v8bf*)(qrow + 32 + half * 8),
                        *(const v8bf*)(qrow + 48 + half * 8));

  float mrun[8], srun[8];
#pragma unroll
  for (int r = 0; r < 8; ++r) { mrun[r] = -INFINITY; srun[r] = 0.f; }

  float*       wbase    = wout + ((size_t)(b * HN + h) * LSEQ + q0) * LSEQ;
  const float* biasbase = bias + ((size_t)h * LSEQ + q0) * LSEQ;
  const int*   maskbase = mask + ((size_t)b * LSEQ + q0) * LSEQ;
  const __bf16* kslab   = kg + (size_t)(b * HN + h) * LSEQ * DK;
  const __bf16* vslab   = vt + (size_t)(b * HN + h) * DK * LSEQ;

  // ---- Phase 1: logits + online softmax stats (K tiles via TDM) ----
  if (wslot == 0) {
    tdm_load_2d_bf16(lds_offset(&kbuf[0][0]), kslab, DK, 16, DK);
    __builtin_amdgcn_s_wait_tensorcnt(0);
  }
  __syncthreads();

  for (int kb = 0; kb < LSEQ / 16; ++kb) {
    const int cur = kb & 1;
    if (wslot == 0 && (kb + 1) < LSEQ / 16)
      tdm_load_2d_bf16(lds_offset(&kbuf[cur ^ 1][0]),
                       kslab + (size_t)(kb + 1) * 16 * DK, DK, 16, DK);

    const int kp0 = kb * 16;
    const __bf16* krow = &kbuf[cur][li * DK];   // B tile from LDS
    const v16bf b0 = cat8(*(const v8bf*)(krow + half * 8),
                          *(const v8bf*)(krow + half * 8 + 16));
    const v16bf b1 = cat8(*(const v8bf*)(krow + 32 + half * 8),
                          *(const v8bf*)(krow + 32 + half * 8 + 16));
    v8f c;
#pragma unroll
    for (int r = 0; r < 8; ++r) c[r] = 0.f;
    c = __builtin_amdgcn_wmma_f32_16x16x32_bf16(false, a0, false, b0, (short)0, c, false, false);
    c = __builtin_amdgcn_wmma_f32_16x16x32_bf16(false, a1, false, b1, (short)0, c, false, false);

#pragma unroll
    for (int r = 0; r < 8; ++r) {
      const int row = r + 8 * half;
      const size_t off = (size_t)row * LSEQ + kp0 + li;
      __builtin_prefetch(biasbase + off + 128, 0, 1);  // 8 tiles ahead
      float l = c[r] * 0.125f;  // 1/sqrt(Dk)
      l = maskbase[off] ? l : -1e30f;
      l += biasbase[off];
      wbase[off] = l;  // raw logit spill into final output slot
      const float mo = mrun[r];
      const float mn = fmaxf(mo, l);
      srun[r] = srun[r] * __expf(mo - mn) + __expf(l - mn);
      mrun[r] = mn;
    }
    if (wslot == 0) __builtin_amdgcn_s_wait_tensorcnt(0);
    __syncthreads();
  }

  // reduce (max,sum) across the 16 lanes that share each row
#pragma unroll
  for (int r = 0; r < 8; ++r) {
    float m = mrun[r], s = srun[r];
#pragma unroll
    for (int o = 1; o < 16; o <<= 1) {
      const float m2 = __shfl_xor(m, o, 32);
      const float s2 = __shfl_xor(s, o, 32);
      const float mn = fmaxf(m, m2);
      s = s * __expf(m - mn) + s2 * __expf(m2 - mn);
      m = mn;
    }
    mrun[r] = m; srun[r] = s;
  }
  if (li == 0) {
#pragma unroll
    for (int r = 0; r < 8; ++r) {
      smax[wslot][half * 8 + r] = mrun[r];
      sinv[wslot][half * 8 + r] = 1.f / srun[r];
    }
  }
  __syncthreads();
  const float mA = smax[wslot][li];  // stats for row li (A-operand row)
  const float iA = sinv[wslot][li];

  // ---- Phase 2: normalize weights (final output) + P @ V^T (V via TDM) ----
  v8f o[4];
#pragma unroll
  for (int t = 0; t < 4; ++t)
#pragma unroll
    for (int r = 0; r < 8; ++r) o[t][r] = 0.f;

  if (wslot == 0) {
    tdm_load_2d_bf16(lds_offset(&vbuf[0][0]), vslab, 32, DK, LSEQ);
    __builtin_amdgcn_s_wait_tensorcnt(0);
  }
  __syncthreads();

  float* lrow = wbase + (size_t)li * LSEQ;
  for (int kb = 0; kb < LSEQ / 32; ++kb) {
    const int cur = kb & 1;
    if (wslot == 0 && (kb + 1) < LSEQ / 32)
      tdm_load_2d_bf16(lds_offset(&vbuf[cur ^ 1][0]),
                       vslab + (size_t)(kb + 1) * 32, 32, DK, LSEQ);

    const int kp0 = kb * 32;
    float la[16];
#pragma unroll
    for (int i = 0; i < 8; ++i) {
      la[i]     = lrow[kp0 + half * 8 + i];
      la[8 + i] = lrow[kp0 + 16 + half * 8 + i];
    }
    v16bf aw;
#pragma unroll
    for (int i = 0; i < 16; ++i) {
      const float pv = __expf(la[i] - mA) * iA;
      la[i] = pv;
      aw[i] = (__bf16)pv;
    }
#pragma unroll
    for (int i = 0; i < 8; ++i) {
      lrow[kp0 + half * 8 + i]      = la[i];
      lrow[kp0 + 16 + half * 8 + i] = la[8 + i];
    }
#pragma unroll
    for (int t = 0; t < 4; ++t) {
      const __bf16* vr = &vbuf[cur][(t * 16 + li) * 32 + half * 8];  // from LDS
      const v16bf bv = cat8(*(const v8bf*)vr, *(const v8bf*)(vr + 16));
      o[t] = __builtin_amdgcn_wmma_f32_16x16x32_bf16(
          false, aw, false, bv, (short)0, o[t], false, false);
    }
    if (wslot == 0) __builtin_amdgcn_s_wait_tensorcnt(0);
    __syncthreads();
  }

#pragma unroll
  for (int t = 0; t < 4; ++t)
#pragma unroll
    for (int r = 0; r < 8; ++r)
      attn[((size_t)(b * LSEQ) + q0 + r + 8 * half) * DMODEL + h * DK + t * 16 + li] =
          (__bf16)o[t][r];
}

// ---------------------------------------------------------------------------
extern "C" void kernel_launch(void* const* d_in, const int* in_sizes, int n_in,
                              void* d_out, int out_size, void* d_ws, size_t ws_size,
                              hipStream_t stream) {
  (void)in_sizes; (void)n_in; (void)out_size; (void)ws_size;

  const float* query = (const float*)d_in[0];
  const float* key   = (const float*)d_in[1];
  const float* value = (const float*)d_in[2];
  const int*   amask = (const int*)d_in[3];
  const float* abias = (const float*)d_in[4];
  const float* Wq = (const float*)d_in[5];
  const float* bq = (const float*)d_in[6];
  const float* Wk = (const float*)d_in[7];
  const float* bk = (const float*)d_in[8];
  const float* Wv = (const float*)d_in[9];
  const float* bv = (const float*)d_in[10];
  const float* Wo = (const float*)d_in[11];
  const float* bo = (const float*)d_in[12];

  const size_t SX = (size_t)BB * LSEQ * DMODEL;  // 4,194,304 elements
  const size_t SW = (size_t)DMODEL * DMODEL;     // 1,048,576 elements

  char* ws = (char*)d_ws;
  size_t off = 0;
  auto take = [&](size_t bytes) {
    void* p = ws + off;
    off = (off + bytes + 255) & ~(size_t)255;
    return p;
  };
  __bf16* Xq  = (__bf16*)take(SX * 2);
  __bf16* Xk  = (__bf16*)take(SX * 2);
  __bf16* Xv  = (__bf16*)take(SX * 2);
  __bf16* Wqb = (__bf16*)take(SW * 2);
  __bf16* Wkb = (__bf16*)take(SW * 2);
  __bf16* Wvb = (__bf16*)take(SW * 2);
  __bf16* Wob = (__bf16*)take(SW * 2);
  __bf16* qb  = (__bf16*)take(SX * 2);  // (B,H,L,Dk)
  __bf16* kb  = (__bf16*)take(SX * 2);  // (B,H,L,Dk)
  __bf16* vtb = (__bf16*)take(SX * 2);  // (B,H,Dk,L)
  __bf16* atb = (__bf16*)take(SX * 2);  // (B,L,D)

  float* out0 = (float*)d_out;                 // (B,L,D)
  float* wout = (float*)d_out + SX;            // (B,H,L,L)

  // 1) f32 -> bf16 conversions
  f32_to_bf16_kernel<<<2048, 256, 0, stream>>>(query, Xq, (int)SX);
  f32_to_bf16_kernel<<<2048, 256, 0, stream>>>(key,   Xk, (int)SX);
  f32_to_bf16_kernel<<<2048, 256, 0, stream>>>(value, Xv, (int)SX);
  f32_to_bf16_kernel<<<1024, 256, 0, stream>>>(Wq, Wqb, (int)SW);
  f32_to_bf16_kernel<<<1024, 256, 0, stream>>>(Wk, Wkb, (int)SW);
  f32_to_bf16_kernel<<<1024, 256, 0, stream>>>(Wv, Wvb, (int)SW);
  f32_to_bf16_kernel<<<1024, 256, 0, stream>>>(Wo, Wob, (int)SW);

  // 2) projections (+RoPE for q,k; transposed store for v)
  proj_kernel<0><<<1024, 128, 0, stream>>>(Xq, Wqb, bq, qb);
  proj_kernel<1><<<1024, 128, 0, stream>>>(Xk, Wkb, bk, kb);
  proj_kernel<2><<<1024, 128, 0, stream>>>(Xv, Wvb, bv, vtb);

  // 3) fused attention (weights written to d_out, context to workspace)
  attn_kernel<<<1024, 128, 0, stream>>>(qb, kb, vtb, amask, abias, wout, atb);

  // 4) output projection -> d_out
  proj_kernel<3><<<1024, 128, 0, stream>>>(atb, Wob, bo, out0);
}
